// BatchLinear_48885317763195
// MI455X (gfx1250) — matrix operand order repaired and checked by reading, not compiled, not run on hardware
//
#include <hip/hip_runtime.h>

// ---------------- problem constants (match reference setup_inputs) ----------
#define G_GRP 64
#define N_TOK 32768
#define D_IN  1024
#define D_OUT 4096
#define CAP_R 768

// ---------------- tiling ----------------------------------------------------
#define TM 128
#define TN 128
#define TK 32
#define KP 40                   // LDS stride in bf16 elems (80B): 16B-aligned frags, conflict-free
#define MT (CAP_R / TM)         // 6 m-tiles per group window
#define NT (D_OUT / TN)         // 32 n-tiles
#define THREADS 256             // 8 wave32

typedef __attribute__((ext_vector_type(16))) __bf16 v16bf;
typedef __attribute__((ext_vector_type(8)))  __bf16 v8bf;
typedef __attribute__((ext_vector_type(8)))  float  v8f;
typedef __attribute__((ext_vector_type(4)))  float  v4f;

// RNE split via native converts: f ~= hi + lo, hi/lo bf16.
__device__ __forceinline__ void split_bf16(float f, __bf16& h, __bf16& l) {
    h = (__bf16)f;              // v_cvt *_bf16_f32 (RNE)
    float fh = (float)h;        // exact widen
    l = (__bf16)(f - fh);
}

// Build a 16-element bf16 fragment from two 16B LDS reads (ds_load_b128).
__device__ __forceinline__ v16bf load_frag2(const __bf16* p0, const __bf16* p1) {
    v8bf x0 = *(const v8bf*)p0;
    v8bf x1 = *(const v8bf*)p1;
    v16bf r;
#pragma unroll
    for (int i = 0; i < 8; ++i) { r[i] = x0[i]; r[8 + i] = x1[i]; }
    return r;
}

struct TileBuf {
    __align__(16) __bf16 Ah[TM * KP];   // A hi [m][k]
    __align__(16) __bf16 Al[TM * KP];   // A lo
    __align__(16) __bf16 Bh[TN * KP];   // B hi [n][k] (transposed)
    __align__(16) __bf16 Bl[TN * KP];   // B lo
};

__global__ void BatchLinear_starts_kernel(const int* __restrict__ counts,
                                          int* __restrict__ starts) {
    if (threadIdx.x == 0) {
        int s = 0;
        for (int e = 0; e < G_GRP; ++e) { starts[e] = s; s += counts[e]; }
    }
}

__global__ __launch_bounds__(THREADS)
void BatchLinear_48885317763195_kernel(const float* __restrict__ x,
                                       const float* __restrict__ w,
                                       const int*   __restrict__ counts,
                                       const int*   __restrict__ starts,
                                       float*       __restrict__ out) {
    const int bid = blockIdx.x;
    const int e   = bid / (MT * NT);
    const int rem = bid % (MT * NT);
    const int mt  = rem / NT;
    const int nt  = rem % NT;

    const int cnt = counts[e];
    const int m0  = mt * TM;            // tile row offset inside the group
    if (m0 >= cnt) return;              // static grid, ragged early-exit
    const int start = starts[e];

    __shared__ TileBuf buf[2];          // 80 KB: double-buffered (320 KB/WGP available)

    const int tid  = threadIdx.x;
    const int lane = tid & 31;
    const int wave = tid >> 5;          // 0..7
    const int wm   = wave & 3;          // 4 m-slots of 32 rows
    const int wn   = wave >> 2;         // 2 n-slots of 64 cols

    v8f acc[2][4];
#pragma unroll
    for (int i = 0; i < 2; ++i)
#pragma unroll
        for (int j = 0; j < 4; ++j)
            acc[i][j] = (v8f){0.f, 0.f, 0.f, 0.f, 0.f, 0.f, 0.f, 0.f};

    // global-load mapping: A tile 128x32 f32, B tile 32x128 f32, float4 each
    const int arow0 = tid >> 3;         // 0..31 (+32 strides)
    const int acol  = (tid & 7) * 4;    // 0..28
    const int bk0   = tid >> 5;         // 0..7  (+8 strides)
    const int bn    = (tid & 31) * 4;   // 0..124

    const float* wptr = w + (size_t)e * D_IN * D_OUT;

    // clamped global rows for this thread's 4 A-row loads
    int growc[4];
#pragma unroll
    for (int i = 0; i < 4; ++i) {
        int g = start + m0 + arow0 + i * 32;
        growc[i] = (g < N_TOK) ? g : (N_TOK - 1);
    }

    const int half = lane >> 4;         // lane half selects K sub-range
    const int idx  = lane & 15;         // A row / B col within 16

    v4f aReg[4], bReg[4];

    // ---- prologue: tile kt=0 -> regs -> bf16 hi/lo -> buf[0] ----
#pragma unroll
    for (int i = 0; i < 4; ++i) {
        aReg[i] = *(const v4f*)(x + (size_t)growc[i] * D_IN + acol);
        bReg[i] = *(const v4f*)(wptr + (size_t)(bk0 + i * 8) * D_OUT + nt * TN + bn);
    }
    {
        TileBuf& S = buf[0];
#pragma unroll
        for (int i = 0; i < 4; ++i) {
            int rr = arow0 + i * 32, kk = bk0 + i * 8;
#pragma unroll
            for (int j = 0; j < 4; ++j) {
                __bf16 h, l;
                split_bf16(aReg[i][j], h, l);
                S.Ah[rr * KP + acol + j] = h;  S.Al[rr * KP + acol + j] = l;
                split_bf16(bReg[i][j], h, l);
                S.Bh[(bn + j) * KP + kk] = h;  S.Bl[(bn + j) * KP + kk] = l;
            }
        }
    }
    __syncthreads();

    int p = 0;
    for (int kt = 0; kt < D_IN; kt += TK) {
        const bool hasNext = (kt + TK < D_IN);

        // ---- issue next tile's global loads first (hidden behind WMMAs) ----
        if (hasNext) {
#pragma unroll
            for (int i = 0; i < 4; ++i) {
                aReg[i] = *(const v4f*)(x + (size_t)growc[i] * D_IN + (kt + TK) + acol);
                bReg[i] = *(const v4f*)(wptr + (size_t)(kt + TK + bk0 + i * 8) * D_OUT + nt * TN + bn);
            }
        }
        if (kt + 2 * TK < D_IN) {       // L2 prefetch two tiles ahead
            __builtin_prefetch(x + (size_t)growc[0] * D_IN + (kt + 2 * TK) + acol, 0, 1);
            __builtin_prefetch(wptr + (size_t)(kt + 2 * TK + bk0) * D_OUT + nt * TN + bn, 0, 1);
        }

        // ---- fragments from buf[p] (ISA VGPR layouts) ----
        // A lane<16: K in {0..7,16..23}; lane>=16: +8
        // B lane<16: K in {0..7,16..23}; lane>=16: +8
        const TileBuf& S = buf[p];
        v16bf ahf[2], alf[2], bhf[4], blf[4];
#pragma unroll
        for (int i = 0; i < 2; ++i) {
            const __bf16* rb = &S.Ah[(wm * 32 + i * 16 + idx) * KP];
            const __bf16* rl = &S.Al[(wm * 32 + i * 16 + idx) * KP];
            ahf[i] = load_frag2(rb + half * 8, rb + half * 8 + 16);
            alf[i] = load_frag2(rl + half * 8, rl + half * 8 + 16);
        }
#pragma unroll
        for (int j = 0; j < 4; ++j) {
            const __bf16* cb = &S.Bh[(wn * 64 + j * 16 + idx) * KP];
            const __bf16* cl = &S.Bl[(wn * 64 + j * 16 + idx) * KP];
            bhf[j] = load_frag2(cb + half * 8, cb + half * 8 + 16);
            blf[j] = load_frag2(cl + half * 8, cl + half * 8 + 16);
        }

        // ---- bf16x3 compensated MMA: x*w ~= xh*wh + xh*wl + xl*wh ----
#pragma unroll
        for (int i = 0; i < 2; ++i)
#pragma unroll
            for (int j = 0; j < 4; ++j) {
                acc[i][j] = __builtin_amdgcn_wmma_f32_16x16x32_bf16(
                    false, ahf[i], false, bhf[j], (short)0, acc[i][j], false, false);
                acc[i][j] = __builtin_amdgcn_wmma_f32_16x16x32_bf16(
                    false, ahf[i], false, blf[j], (short)0, acc[i][j], false, false);
                acc[i][j] = __builtin_amdgcn_wmma_f32_16x16x32_bf16(
                    false, alf[i], false, bhf[j], (short)0, acc[i][j], false, false);
            }

        // ---- convert + store next tile into the other buffer, one barrier ----
        if (hasNext) {
            TileBuf& D = buf[p ^ 1];
#pragma unroll
            for (int i = 0; i < 4; ++i) {
                int rr = arow0 + i * 32, kk = bk0 + i * 8;
#pragma unroll
                for (int j = 0; j < 4; ++j) {
                    __bf16 h, l;
                    split_bf16(aReg[i][j], h, l);
                    D.Ah[rr * KP + acol + j] = h;  D.Al[rr * KP + acol + j] = l;
                    split_bf16(bReg[i][j], h, l);
                    D.Bh[(bn + j) * KP + kk] = h;  D.Bl[(bn + j) * KP + kk] = l;
                }
            }
            __syncthreads();
            p ^= 1;
        }
    }

    // ---- epilogue: C/D layout — VGPR v: lane<16 -> M=v, lane>=16 -> M=v+8 ----
#pragma unroll
    for (int i = 0; i < 2; ++i) {
#pragma unroll
        for (int v = 0; v < 8; ++v) {
            int mrow = m0 + wm * 32 + i * 16 + half * 8 + v;  // row inside group
            if (mrow < cnt) {
                size_t grow = (size_t)(start + mrow);
#pragma unroll
                for (int j = 0; j < 4; ++j)
                    out[grow * D_OUT + nt * TN + wn * 64 + j * 16 + idx] = acc[i][j][v];
            }
        }
    }
}

extern "C" void kernel_launch(void* const* d_in, const int* in_sizes, int n_in,
                              void* d_out, int out_size, void* d_ws, size_t ws_size,
                              hipStream_t stream) {
    const float* x      = (const float*)d_in[0];
    const float* weight = (const float*)d_in[1];
    const int*   counts = (const int*)d_in[2];
    float*       out    = (float*)d_out;
    int*         starts = (int*)d_ws;           // G ints of scratch

    BatchLinear_starts_kernel<<<1, 32, 0, stream>>>(counts, starts);

    dim3 grid(G_GRP * MT * NT);                 // 12288 blocks, static, ragged early-exit
    BatchLinear_48885317763195_kernel<<<grid, THREADS, 0, stream>>>(
        x, weight, counts, starts, out);
}
